// AttentionFusion_80083960201743
// MI455X (gfx1250) — matrix operand order repaired and checked by reading, not compiled, not run on hardware
//
#include <hip/hip_runtime.h>

typedef _Float16 h16v __attribute__((ext_vector_type(16)));
typedef float    f8v  __attribute__((ext_vector_type(8)));

#define C_DIM 256
#define N_TOK 8192
#define HW_   4096
#define BATCH 2

__device__ __forceinline__ f8v wmma_f16(h16v a, h16v b, f8v c) {
  return __builtin_amdgcn_wmma_f32_16x16x32_f16(false, a, false, b, (short)0, c,
                                                false, false);
}

// A-fragment: 16x32 f16 tile (M x K), row-major source, stride in halves.
__device__ __forceinline__ h16v frag_a(const _Float16* base, int stride_h) {
  int l = threadIdx.x & 31;
  const _Float16* p = base + (l & 15) * stride_h + ((l >> 4) * 8);
  h16v r;
  ((float4*)&r)[0] = *(const float4*)(p);
  ((float4*)&r)[1] = *(const float4*)(p + 16);
  return r;
}

// B-fragment for D = A * B^T form: source is B^T row-major (N x K), stride in halves.
__device__ __forceinline__ h16v frag_b(const _Float16* base, int stride_h) {
  int l = threadIdx.x & 31;
  const _Float16* p = base + (l & 15) * stride_h + ((l >> 4) * 8);
  h16v r;
  ((float4*)&r)[0] = *(const float4*)(p);
  ((float4*)&r)[1] = *(const float4*)(p + 16);
  return r;
}

__device__ __forceinline__ float rmax16(float x) {
  x = fmaxf(x, __shfl_xor(x, 1, 32));
  x = fmaxf(x, __shfl_xor(x, 2, 32));
  x = fmaxf(x, __shfl_xor(x, 4, 32));
  x = fmaxf(x, __shfl_xor(x, 8, 32));
  return x;
}
__device__ __forceinline__ float rsum16(float x) {
  x += __shfl_xor(x, 1, 32);
  x += __shfl_xor(x, 2, 32);
  x += __shfl_xor(x, 4, 32);
  x += __shfl_xor(x, 8, 32);
  return x;
}

// ---- gfx1250 async global->LDS DMA (ASYNCcnt-tracked) ----
__device__ __forceinline__ void async_ld16(unsigned lds, const void* g) {
  asm volatile("global_load_async_to_lds_b128 %0, %1, off"
               :: "v"(lds), "v"(g) : "memory");
}
__device__ __forceinline__ void wait_async0() {
  asm volatile("s_wait_asynccnt 0x0" ::: "memory");
}
__device__ __forceinline__ void wait_async16() {
  asm volatile("s_wait_asynccnt 0x10" ::: "memory");
}

// ---------- kernel 1: weights f32 -> f16 ----------
__global__ void __launch_bounds__(256)
k_wconv(const float* __restrict__ wq, const float* __restrict__ wk,
        const float* __restrict__ wv, const float* __restrict__ wo,
        _Float16* q16, _Float16* k16, _Float16* v16, _Float16* o16) {
  int i = blockIdx.x * 256 + threadIdx.x;
  if (i < C_DIM * C_DIM) {
    q16[i] = (_Float16)wq[i];
    k16[i] = (_Float16)wk[i];
    v16[i] = (_Float16)wv[i];
    o16[i] = (_Float16)wo[i];
  }
}

// ---------- kernel 2: X[b][n][c] = feat_m[b][c][hw] in f16 ----------
__global__ void __launch_bounds__(256)
k_buildx(const float* __restrict__ f0, const float* __restrict__ f1,
         _Float16* __restrict__ X) {
  int i = blockIdx.x * 256 + threadIdx.x;    // consecutive threads -> consecutive hw
  int hw = i & (HW_ - 1);
  int c  = (i >> 12) & (C_DIM - 1);
  int m  = (i >> 20) & 1;
  int b  = i >> 21;
  const float* src = m ? f1 : f0;
  float v = src[((b * C_DIM + c) << 12) + hw];
  int n = m * HW_ + hw;
  X[(b * N_TOK + n) * C_DIM + c] = (_Float16)v;
}

// ---------- kernel 3: Q/K/V projections (Q,K row-major f16; V transposed) ----------
__global__ void __launch_bounds__(128)
k_qkv(const _Float16* __restrict__ X,
      const _Float16* __restrict__ Wq16, const float* __restrict__ bq,
      const _Float16* __restrict__ Wk16, const float* __restrict__ bk,
      const _Float16* __restrict__ Wv16, const float* __restrict__ bv,
      _Float16* __restrict__ Q, _Float16* __restrict__ K,
      _Float16* __restrict__ Vt) {
  int gw  = blockIdx.x * 4 + (threadIdx.x >> 5);   // 0..1023 global waves
  int nb  = gw * 16;                               // 16 rows per wave
  int b   = nb >> 13;
  int nn  = nb & (N_TOK - 1);
  int l   = threadIdx.x & 31;
  int col = l & 15, kh = l >> 4;

  h16v af[8];
#pragma unroll
  for (int kc = 0; kc < 8; ++kc)
    af[kc] = frag_a(X + nb * C_DIM + kc * 32, C_DIM);

  const _Float16* Ws[3] = {Wq16, Wk16, Wv16};
  const float*    bs[3] = {bq, bk, bv};
  for (int s = 0; s < 3; ++s) {
    const _Float16* W    = Ws[s];
    const float*    bias = bs[s];
    for (int ct = 0; ct < 16; ++ct) {
      float bvv = bias[ct * 16 + col];
      // preload all B-fragments (one load clause), then a WMMA burst
      h16v bf[8];
#pragma unroll
      for (int kc = 0; kc < 8; ++kc)
        bf[kc] = frag_b(W + (ct * 16) * C_DIM + kc * 32, C_DIM);
      f8v acc;
#pragma unroll
      for (int v = 0; v < 8; ++v) acc[v] = bvv;
#pragma unroll
      for (int kc = 0; kc < 8; ++kc) acc = wmma_f16(af[kc], bf[kc], acc);
      if (s < 2) {
        _Float16* out = (s == 0) ? Q : K;
#pragma unroll
        for (int v = 0; v < 8; ++v)
          out[(nb + v + 8 * kh) * C_DIM + ct * 16 + col] = (_Float16)acc[v];
      } else {
#pragma unroll
        for (int v = 0; v < 8; ++v)
          Vt[(b * C_DIM + ct * 16 + col) * N_TOK + nn + v + 8 * kh] = (_Float16)acc[v];
      }
    }
  }
}

// ---------- kernel 4: flash attention with async double-buffered K/V tiles ----------
__global__ void __launch_bounds__(128)
k_flash(const _Float16* __restrict__ Q, const _Float16* __restrict__ K,
        const _Float16* __restrict__ Vt, _Float16* __restrict__ O) {
  __shared__ __align__(16) _Float16 sK[2][32 * C_DIM];   // 2 x 16 KB
  __shared__ __align__(16) _Float16 sV[2][C_DIM * 32];   // 2 x 16 KB
  __shared__ __align__(16) _Float16 sP[4][16 * 32];      // 4 KB

  int b   = blockIdx.x >> 7;
  int q0  = (blockIdx.x & 127) * 64;
  int w   = threadIdx.x >> 5;
  int l   = threadIdx.x & 31;
  int col = l & 15, kh = l >> 4;
  int tid = threadIdx.x;
  int qrow = q0 + w * 16;

  unsigned ldsK[2] = {(unsigned)(uintptr_t)&sK[0][0], (unsigned)(uintptr_t)&sK[1][0]};
  unsigned ldsV[2] = {(unsigned)(uintptr_t)&sV[0][0], (unsigned)(uintptr_t)&sV[1][0]};

  // issue one 32-key K/V tile as 16 async b128 DMAs per thread
  auto issue_tile = [&](int buf, int k0) {
    const char* gk = (const char*)(K + (size_t)(b * N_TOK + k0) * C_DIM);
#pragma unroll
    for (int i = 0; i < 8; ++i) {
      int e = tid + i * 128;                 // 0..1023 float4 slots
      async_ld16(ldsK[buf] + e * 16, gk + (size_t)e * 16);
    }
#pragma unroll
    for (int i = 0; i < 8; ++i) {
      int e = tid + i * 128;
      int c = e >> 2, j = e & 3;
      const char* gv =
          (const char*)(Vt + (size_t)(b * C_DIM + c) * N_TOK + k0) + j * 16;
      async_ld16(ldsV[buf] + e * 16, gv);
    }
  };

  h16v qa[8];
#pragma unroll
  for (int kc = 0; kc < 8; ++kc)
    qa[kc] = frag_a(Q + (b * N_TOK + qrow) * C_DIM + kc * 32, C_DIM);

  f8v o[16];
#pragma unroll
  for (int t = 0; t < 16; ++t)
#pragma unroll
    for (int v = 0; v < 8; ++v) o[t][v] = 0.f;
  float mrow[8], lrow[8];
#pragma unroll
  for (int v = 0; v < 8; ++v) { mrow[v] = -3.0e38f; lrow[v] = 0.f; }

  const int NITER = N_TOK / 32;
  issue_tile(0, 0);
  for (int kb = 0; kb < NITER; ++kb) {
    int p = kb & 1;
    if (kb + 1 < NITER) {
      issue_tile(1 - p, (kb + 1) * 32);   // prefetch next tile (prev read done at
      wait_async16();                     //  last iteration's trailing barrier)
    } else {
      wait_async0();
    }
    __syncthreads();                      // tile p complete & visible to all waves

    const _Float16* cK = sK[p];
    const _Float16* cV = sV[p];

    // S = Q * K^T  (two 16-key subtiles)
    f8v s0, s1;
#pragma unroll
    for (int v = 0; v < 8; ++v) { s0[v] = 0.f; s1[v] = 0.f; }
#pragma unroll
    for (int kc = 0; kc < 8; ++kc) {
      s0 = wmma_f16(qa[kc], frag_b(cK + kc * 32, C_DIM), s0);
      s1 = wmma_f16(qa[kc], frag_b(cK + 16 * C_DIM + kc * 32, C_DIM), s1);
    }

    // online softmax (scale = 1/sqrt(256) = 1/16)
    float p0[8], p1[8];
#pragma unroll
    for (int v = 0; v < 8; ++v) {
      float a  = s0[v] * 0.0625f;
      float bb = s1[v] * 0.0625f;
      float tm = rmax16(fmaxf(a, bb));
      float nm = fmaxf(mrow[v], tm);
      float corr = __expf(mrow[v] - nm);
      p0[v] = __expf(a - nm);
      p1[v] = __expf(bb - nm);
      float rs = rsum16(p0[v] + p1[v]);
      lrow[v] = lrow[v] * corr + rs;
      mrow[v] = nm;
#pragma unroll
      for (int t = 0; t < 16; ++t) o[t][v] *= corr;
    }

    // stage P (C/D layout -> LDS) then reload as A-fragment
    _Float16* pw = sP[w];
#pragma unroll
    for (int v = 0; v < 8; ++v) {
      pw[(v + 8 * kh) * 32 + col]      = (_Float16)p0[v];
      pw[(v + 8 * kh) * 32 + 16 + col] = (_Float16)p1[v];
    }
    __asm__ volatile("" ::: "memory");   // keep f16 stores before punned loads
    h16v pa = frag_a(pw, 32);

    // O += P * V   (Vt tile gives B^T row-major directly)
#pragma unroll
    for (int t = 0; t < 16; ++t)
      o[t] = wmma_f16(pa, frag_b(cV + (t * 16) * 32, 32), o[t]);

    __syncthreads();                      // all waves done with tile p
  }

#pragma unroll
  for (int v = 0; v < 8; ++v) {
    float inv = 1.0f / lrow[v];
#pragma unroll
    for (int t = 0; t < 16; ++t) o[t][v] *= inv;
  }
#pragma unroll
  for (int t = 0; t < 16; ++t)
#pragma unroll
    for (int v = 0; v < 8; ++v)
      O[(b * N_TOK + qrow + v + 8 * kh) * C_DIM + t * 16 + col] = (_Float16)o[t][v];
}

// ---------- kernel 5: mean over modalities + output projection + NCHW store ----------
__global__ void __launch_bounds__(128)
k_oproj(const _Float16* __restrict__ O, const _Float16* __restrict__ Wo16,
        const float* __restrict__ bo, float* __restrict__ out) {
  int gw  = blockIdx.x * 4 + (threadIdx.x >> 5);  // 0..511
  int b   = gw >> 8;
  int hw0 = (gw & 255) * 16;
  int l   = threadIdx.x & 31;
  int col = l & 15, kh = l >> 4;

  h16v af[8];
#pragma unroll
  for (int kc = 0; kc < 8; ++kc) {
    h16v a0 = frag_a(O + (b * N_TOK + hw0) * C_DIM + kc * 32, C_DIM);
    h16v a1 = frag_a(O + (b * N_TOK + HW_ + hw0) * C_DIM + kc * 32, C_DIM);
    h16v r;
#pragma unroll
    for (int e = 0; e < 16; ++e)
      r[e] = (_Float16)(((float)a0[e] + (float)a1[e]) * 0.5f);
    af[kc] = r;
  }
  for (int ct = 0; ct < 16; ++ct) {
    float bvv = bo[ct * 16 + col];
    h16v bf[8];
#pragma unroll
    for (int kc = 0; kc < 8; ++kc)
      bf[kc] = frag_b(Wo16 + (ct * 16) * C_DIM + kc * 32, C_DIM);
    f8v acc;
#pragma unroll
    for (int v = 0; v < 8; ++v) acc[v] = bvv;
#pragma unroll
    for (int kc = 0; kc < 8; ++kc) acc = wmma_f16(af[kc], bf[kc], acc);
#pragma unroll
    for (int v = 0; v < 8; ++v)
      out[(b * C_DIM + ct * 16 + col) * HW_ + hw0 + v + 8 * kh] = acc[v];
  }
}

extern "C" void kernel_launch(void* const* d_in, const int* in_sizes, int n_in,
                              void* d_out, int out_size, void* d_ws, size_t ws_size,
                              hipStream_t stream) {
  const float* f0 = (const float*)d_in[0];
  const float* f1 = (const float*)d_in[1];
  const float* Wq = (const float*)d_in[2];
  const float* bq = (const float*)d_in[3];
  const float* Wk = (const float*)d_in[4];
  const float* bk = (const float*)d_in[5];
  const float* Wv = (const float*)d_in[6];
  const float* bv = (const float*)d_in[7];
  const float* Wo = (const float*)d_in[8];
  const float* bo = (const float*)d_in[9];

  char* ws = (char*)d_ws;
  const size_t szNC = (size_t)BATCH * N_TOK * C_DIM * sizeof(_Float16); // 8 MiB
  _Float16* X    = (_Float16*)(ws);
  _Float16* Qh   = (_Float16*)(ws + 1 * szNC);
  _Float16* Kh   = (_Float16*)(ws + 2 * szNC);
  _Float16* Vt   = (_Float16*)(ws + 3 * szNC);
  _Float16* Oh   = (_Float16*)(ws + 4 * szNC);
  _Float16* Wq16 = (_Float16*)(ws + 5 * szNC);
  _Float16* Wk16 = Wq16 + C_DIM * C_DIM;
  _Float16* Wv16 = Wk16 + C_DIM * C_DIM;
  _Float16* Wo16 = Wv16 + C_DIM * C_DIM;

  k_wconv<<<256, 256, 0, stream>>>(Wq, Wk, Wv, Wo, Wq16, Wk16, Wv16, Wo16);
  k_buildx<<<(BATCH * N_TOK * C_DIM) / 256, 256, 0, stream>>>(f0, f1, X);
  k_qkv<<<256, 128, 0, stream>>>(X, Wq16, bq, Wk16, bk, Wv16, bv, Qh, Kh, Vt);
  k_flash<<<256, 128, 0, stream>>>(Qh, Kh, Vt, Oh);
  k_oproj<<<128, 128, 0, stream>>>(Oh, Wo16, bo, (float*)d_out);
}
